// GPT_29429115912988
// MI455X (gfx1250) — matrix operand order repaired and checked by reading, not compiled, not run on hardware
//
#include <hip/hip_runtime.h>
#include <hip/hip_bf16.h>
#include <math.h>

// ---------------- model dims ----------------
#define Bb 2
#define Tt 1024
#define Dd 768
#define Hh 12
#define Ee 2
#define Ll 2
#define Vv 50257
#define HDd 64
#define NN (Bb*Tt)
#define FF (4*Dd)

// ---------------- vector types ----------------
typedef __attribute__((ext_vector_type(16))) __bf16 v16bf;
typedef __attribute__((ext_vector_type(8)))  __bf16 v8bf;
typedef __attribute__((ext_vector_type(8)))  float  v8f;

__device__ __forceinline__ v16bf join16(v8bf a, v8bf b) {
  return __builtin_shufflevector(a, b, 0,1,2,3,4,5,6,7,8,9,10,11,12,13,14,15);
}

// ---------------- WMMA GEMM ----------------
// C[M,N] = A[M,K] (f32 row-major) x B (f32), bf16 tiles in LDS, f32 accum.
// flags: 1=transB (B is [N,K]), 2=bias, 4=gelu, 8=accumulate into C.
#define BM 64
#define BN 128
#define BK 32
#define LDP 8
enum { FLAG_TB = 1, FLAG_BIAS = 2, FLAG_GELU = 4, FLAG_ACC = 8 };

__global__ __launch_bounds__(256) void wmma_gemm(
    const float* __restrict__ A, const float* __restrict__ Bm,
    const float* __restrict__ bias, float* __restrict__ C,
    int M, int Nn, int K, int flags)
{
  __shared__ __align__(16) __bf16 As[BM][BK + LDP];
  __shared__ __align__(16) __bf16 Bs[BN][BK + LDP];

  const int tid  = threadIdx.x;
  const int lane = tid & 31;
  const int w    = tid >> 5;
  const int hf   = lane >> 4;     // half-wave: 0 or 1
  const int l15  = lane & 15;
  const int wm   = w >> 2, wn = w & 3;
  const int m0   = blockIdx.y * BM;
  const int n0   = blockIdx.x * BN;

  v8f acc[2][2];
  for (int i = 0; i < 2; i++)
    for (int j = 0; j < 2; j++)
      acc[i][j] = v8f{};

  const int arow = tid >> 2;            // 0..63
  const int acol = (tid & 3) * 8;       // 0,8,16,24
  const bool tb  = (flags & FLAG_TB);
  const int brow_t = tid >> 1;          // transB: 0..127
  const int bk_t   = (tid & 1) * 16;
  const int bkr    = tid >> 3;          // !transB: 0..31
  const int bnc    = (tid & 7) * 16;

  for (int k0 = 0; k0 < K; k0 += BK) {
    __syncthreads();
    // ---- stage A tile (convert f32 -> bf16) ----
    {
      int gm = m0 + arow;
      const float* ap = A + (size_t)gm * K + k0 + acol;
      if (gm < M) {
        #pragma unroll
        for (int j = 0; j < 8; j++) As[arow][acol + j] = (__bf16)ap[j];
        if (k0 + BK < K) __builtin_prefetch(ap + BK);   // global_prefetch_b8
      } else {
        #pragma unroll
        for (int j = 0; j < 8; j++) As[arow][acol + j] = (__bf16)0.f;
      }
    }
    // ---- stage B tile, stored transposed as Bs[n][k] ----
    if (tb) {
      int gn = n0 + brow_t;
      const float* bp = Bm + (size_t)gn * K + k0 + bk_t;
      if (gn < Nn) {
        #pragma unroll
        for (int j = 0; j < 16; j++) Bs[brow_t][bk_t + j] = (__bf16)bp[j];
        if (k0 + BK < K) __builtin_prefetch(bp + BK);
      } else {
        #pragma unroll
        for (int j = 0; j < 16; j++) Bs[brow_t][bk_t + j] = (__bf16)0.f;
      }
    } else {
      const float* bp = Bm + (size_t)(k0 + bkr) * Nn + n0 + bnc;
      #pragma unroll
      for (int j = 0; j < 16; j++) {
        int gn = n0 + bnc + j;
        float vv = (gn < Nn) ? bp[j] : 0.f;
        Bs[bnc + j][bkr] = (__bf16)vv;
      }
    }
    __syncthreads();

    // ---- fragments per ISA layouts ----
    v16bf af[2], bfr[2];
    #pragma unroll
    for (int mi = 0; mi < 2; mi++) {
      int r = wm * 32 + mi * 16 + l15;
      af[mi] = join16(*(const v8bf*)&As[r][hf * 8],
                      *(const v8bf*)&As[r][16 + hf * 8]);
    }
    #pragma unroll
    for (int ni = 0; ni < 2; ni++) {
      int c = wn * 32 + ni * 16 + l15;
      bfr[ni] = join16(*(const v8bf*)&Bs[c][hf * 8],
                       *(const v8bf*)&Bs[c][16 + hf * 8]);
    }
    #pragma unroll
    for (int mi = 0; mi < 2; mi++)
      #pragma unroll
      for (int ni = 0; ni < 2; ni++)
        acc[mi][ni] = __builtin_amdgcn_wmma_f32_16x16x32_bf16(
            false, af[mi], false, bfr[ni], (short)0, acc[mi][ni], false, false);
  }

  // ---- epilogue: bias / gelu / accumulate ----
  #pragma unroll
  for (int mi = 0; mi < 2; mi++) {
    #pragma unroll
    for (int ni = 0; ni < 2; ni++) {
      #pragma unroll
      for (int i = 0; i < 8; i++) {
        int r = m0 + wm * 32 + mi * 16 + hf * 8 + i;
        int c = n0 + wn * 32 + ni * 16 + l15;
        if (r < M && c < Nn) {
          float vv = acc[mi][ni][i];
          if (flags & FLAG_BIAS) vv += bias[c];
          if (flags & FLAG_GELU) {
            float u = vv;
            vv = 0.5f * u * (1.f + tanhf(0.7978845608028654f *
                                         (u + 0.044715f * u * u * u)));
          }
          size_t o = (size_t)r * Nn + c;
          if (flags & FLAG_ACC) C[o] += vv; else C[o] = vv;
        }
      }
    }
  }
}

// ---------------- flash attention (per expert) ----------------
// q,k,v,o are [N, D] with head slice at column h*64. 4 waves/block,
// each wave owns 16 query rows; key chunks of 32 with online softmax.
__global__ __launch_bounds__(128) void moa_attn_k(
    const float* __restrict__ q, const float* __restrict__ k,
    const float* __restrict__ v, float* __restrict__ o)
{
  __shared__ __align__(16) __bf16 Pb[4][16][BK + LDP];   // 16x32 P per wave
  __shared__ __align__(16) __bf16 Vt[4][64][BK + LDP];   // V chunk transposed

  const int tid  = threadIdx.x;
  const int w    = tid >> 5, lane = tid & 31;
  const int hf   = lane >> 4, l15 = lane & 15;
  const int bh   = blockIdx.y;
  const int b    = bh / Hh, h = bh % Hh;
  const size_t base = (size_t)b * Tt * Dd + (size_t)h * HDd;
  const int q0   = blockIdx.x * 64 + w * 16;

  // Q fragments: two K-steps of 32 along head dim (HD=64)
  v16bf qf[2];
  {
    const float* qp = q + base + (size_t)(q0 + l15) * Dd;
    #pragma unroll
    for (int s = 0; s < 2; s++) {
      v16bf f;
      #pragma unroll
      for (int j = 0; j < 8; j++) {
        f[j]     = (__bf16)qp[s * 32 + hf * 8 + j];
        f[8 + j] = (__bf16)qp[s * 32 + 16 + hf * 8 + j];
      }
      qf[s] = f;
    }
  }

  float rmax[8], rsum[8];
  v8f oacc[4];
  #pragma unroll
  for (int i = 0; i < 8; i++) { rmax[i] = -3.0e38f; rsum[i] = 0.f; }
  #pragma unroll
  for (int a = 0; a < 4; a++) oacc[a] = v8f{};

  const int qend = q0 + 15;
  for (int kb = 0; kb <= qend; kb += 32) {
    // stage V chunk transposed: Vt[d][kk]
    {
      const float* vp = v + base + (size_t)(kb + lane) * Dd;
      #pragma unroll 8
      for (int d = 0; d < 64; d++) Vt[w][d][lane] = (__bf16)vp[d];
    }
    // S = Q @ K^T for two 16-key groups (K fragments straight from global)
    v8f sacc[2];
    #pragma unroll
    for (int ni = 0; ni < 2; ni++) {
      sacc[ni] = v8f{};
      const float* kp = k + base + (size_t)(kb + ni * 16 + l15) * Dd;
      #pragma unroll
      for (int s = 0; s < 2; s++) {
        v16bf kf;
        #pragma unroll
        for (int j = 0; j < 16; j++) kf[j] = (__bf16)kp[s * 32 + (j < 8 ? hf * 8 + j : 16 + hf * 8 + (j - 8))];
        sacc[ni] = __builtin_amdgcn_wmma_f32_16x16x32_bf16(
            false, qf[s], false, kf, (short)0, sacc[ni], false, false);
      }
    }
    // online softmax (row lives in a 16-lane group; reduce with shfl_xor)
    float corr[8], p0v[8], p1v[8];
    #pragma unroll
    for (int i = 0; i < 8; i++) {
      int m  = i + 8 * hf;
      int qi = q0 + m;
      float s0 = sacc[0][i] * 0.125f; if (kb + l15 > qi)       s0 = -1e30f;
      float s1 = sacc[1][i] * 0.125f; if (kb + 16 + l15 > qi)  s1 = -1e30f;
      float mx = fmaxf(s0, s1);
      for (int off = 1; off < 16; off <<= 1) mx = fmaxf(mx, __shfl_xor(mx, off));
      float mnew = fmaxf(rmax[i], mx);
      float c  = expf(rmax[i] - mnew);
      float p0 = expf(s0 - mnew), p1 = expf(s1 - mnew);
      float ps = p0 + p1;
      for (int off = 1; off < 16; off <<= 1) ps += __shfl_xor(ps, off);
      rsum[i] = rsum[i] * c + ps;
      rmax[i] = mnew;
      corr[i] = c; p0v[i] = p0; p1v[i] = p1;
    }
    #pragma unroll
    for (int a = 0; a < 4; a++)
      #pragma unroll
      for (int i = 0; i < 8; i++) oacc[a][i] *= corr[i];
    // P -> LDS (C-layout -> A-layout transpose through LDS)
    #pragma unroll
    for (int i = 0; i < 8; i++) {
      Pb[w][i + 8 * hf][l15]      = (__bf16)p0v[i];
      Pb[w][i + 8 * hf][16 + l15] = (__bf16)p1v[i];
    }
    asm volatile("s_wait_dscnt 0" ::: "memory");
    v16bf pf = join16(*(const v8bf*)&Pb[w][l15][hf * 8],
                      *(const v8bf*)&Pb[w][l15][16 + hf * 8]);
    #pragma unroll
    for (int a = 0; a < 4; a++) {
      v16bf vf = join16(*(const v8bf*)&Vt[w][a * 16 + l15][hf * 8],
                        *(const v8bf*)&Vt[w][a * 16 + l15][16 + hf * 8]);
      oacc[a] = __builtin_amdgcn_wmma_f32_16x16x32_bf16(
          false, pf, false, vf, (short)0, oacc[a], false, false);
    }
  }
  // write O
  #pragma unroll
  for (int a = 0; a < 4; a++) {
    #pragma unroll
    for (int i = 0; i < 8; i++) {
      int m = i + 8 * hf;
      o[base + (size_t)(q0 + m) * Dd + a * 16 + l15] = oacc[a][i] / rsum[i];
    }
  }
}

// ---------------- LayerNorm ----------------
__global__ __launch_bounds__(256) void layernorm_k(
    const float* __restrict__ x, const float* __restrict__ g,
    const float* __restrict__ bta, float* __restrict__ y)
{
  int row = blockIdx.x;
  int tid = threadIdx.x;
  __shared__ float redS[8], redQ[8];
  __shared__ float stats[2];
  const float* xr = x + (size_t)row * Dd;
  float s = 0.f, sq = 0.f;
  for (int d = tid; d < Dd; d += 256) { float v = xr[d]; s += v; sq += v * v; }
  for (int off = 16; off > 0; off >>= 1) {
    s += __shfl_xor(s, off); sq += __shfl_xor(sq, off);
  }
  if ((tid & 31) == 0) { redS[tid >> 5] = s; redQ[tid >> 5] = sq; }
  __syncthreads();
  if (tid == 0) {
    float S = 0, Q = 0;
    for (int i = 0; i < 8; i++) { S += redS[i]; Q += redQ[i]; }
    float mean = S / Dd;
    float var  = Q / Dd - mean * mean;
    stats[0] = mean; stats[1] = rsqrtf(var + 1e-5f);
  }
  __syncthreads();
  float mean = stats[0], rstd = stats[1];
  for (int d = tid; d < Dd; d += 256)
    y[(size_t)row * Dd + d] = (xr[d] - mean) * rstd * g[d] + bta[d];
}

// ---------------- embedding ----------------
__global__ void embed_k(const int* __restrict__ idx, const float* __restrict__ wte,
                        const float* __restrict__ wpe, float* __restrict__ x)
{
  int i = blockIdx.x * blockDim.x + threadIdx.x;
  if (i >= NN * Dd) return;
  int n = i / Dd, d = i - n * Dd;
  int t = n % Tt;
  x[i] = wte[(size_t)idx[n] * Dd + d] + wpe[(size_t)t * Dd + d];
}

// ---------------- router: coef = softmax(gate + noise*softplus(noise_gate)) ----------------
__global__ __launch_bounds__(32) void router_k(
    const float* __restrict__ lnx, const float* __restrict__ gw,
    const float* __restrict__ nw, const float* __restrict__ noise,
    float* __restrict__ coef)
{
  int row = blockIdx.x; int lane = threadIdx.x;
  const float* xr = lnx + (size_t)row * Dd;
  float g0 = 0, g1 = 0, s0 = 0, s1 = 0;
  for (int d = lane; d < Dd; d += 32) {
    float v = xr[d];
    g0 += v * gw[d * Ee + 0]; g1 += v * gw[d * Ee + 1];
    s0 += v * nw[d * Ee + 0]; s1 += v * nw[d * Ee + 1];
  }
  for (int off = 16; off > 0; off >>= 1) {
    g0 += __shfl_xor(g0, off); g1 += __shfl_xor(g1, off);
    s0 += __shfl_xor(s0, off); s1 += __shfl_xor(s1, off);
  }
  if (lane == 0) {
    float sp0 = (s0 > 20.f) ? s0 : log1pf(expf(s0));
    float sp1 = (s1 > 20.f) ? s1 : log1pf(expf(s1));
    float l0 = g0 + noise[(size_t)row * Ee + 0] * sp0;
    float l1 = g1 + noise[(size_t)row * Ee + 1] * sp1;
    float m  = fmaxf(l0, l1);
    float e0 = expf(l0 - m), e1 = expf(l1 - m);
    float inv = 1.f / (e0 + e1);
    coef[(size_t)row * Ee + 0] = e0 * inv;
    coef[(size_t)row * Ee + 1] = e1 * inv;
  }
}

// ---------------- expert combine: x += coef[:,e] * tmp ----------------
__global__ void scale_add_k(float* __restrict__ x, const float* __restrict__ t,
                            const float* __restrict__ coef, int e)
{
  int i = blockIdx.x * blockDim.x + threadIdx.x;
  if (i >= NN * Dd) return;
  int n = i / Dd;
  x[i] += coef[(size_t)n * Ee + e] * t[i];
}

// ---------------- host orchestration ----------------
extern "C" void kernel_launch(void* const* d_in, const int* in_sizes, int n_in,
                              void* d_out, int out_size, void* d_ws, size_t ws_size,
                              hipStream_t stream)
{
  (void)in_sizes; (void)n_in; (void)out_size; (void)ws_size;
  const int*   idx     = (const int*)  d_in[0];
  const float* wte     = (const float*)d_in[1];
  const float* wpe     = (const float*)d_in[2];
  const float* ln1_g   = (const float*)d_in[3];
  const float* ln1_b   = (const float*)d_in[4];
  const float* ln2_g   = (const float*)d_in[5];
  const float* ln2_b   = (const float*)d_in[6];
  const float* lnf_g   = (const float*)d_in[7];
  const float* lnf_b   = (const float*)d_in[8];
  const float* Wk      = (const float*)d_in[9];
  const float* Wv      = (const float*)d_in[10];
  const float* Wq      = (const float*)d_in[11];
  const float* Wo      = (const float*)d_in[12];
  const float* gate_w  = (const float*)d_in[13];
  const float* noise_w = (const float*)d_in[14];
  const float* fc_w    = (const float*)d_in[15];
  const float* fc_b    = (const float*)d_in[16];
  const float* proj_w  = (const float*)d_in[17];
  const float* proj_b  = (const float*)d_in[18];
  const float* noise   = (const float*)d_in[19];
  float* out = (float*)d_out;

  float* x    = (float*)d_ws;
  float* lnx  = x    + (size_t)NN * Dd;
  float* kbuf = lnx  + (size_t)NN * Dd;
  float* vbuf = kbuf + (size_t)NN * Dd;
  float* qbuf = vbuf + (size_t)NN * Dd;
  float* obuf = qbuf + (size_t)NN * Dd;
  float* coef = obuf + (size_t)NN * Dd;
  float* tmp  = coef + (size_t)NN * Ee;   // NN*FF floats

  dim3 blk256(256), blk128(128), blk32(32);

  embed_k<<<(NN * Dd + 255) / 256, blk256, 0, stream>>>(idx, wte, wpe, x);

  for (int l = 0; l < Ll; l++) {
    layernorm_k<<<NN, blk256, 0, stream>>>(x, ln1_g + l * Dd, ln1_b + l * Dd, lnx);
    router_k<<<NN, blk32, 0, stream>>>(lnx, gate_w + (size_t)l * Dd * Ee,
                                       noise_w + (size_t)l * Dd * Ee,
                                       noise + (size_t)l * NN * Ee, coef);
    dim3 gD((Dd + BN - 1) / BN, (NN + BM - 1) / BM);
    wmma_gemm<<<gD, blk256, 0, stream>>>(lnx, Wk + (size_t)l * Dd * Dd, nullptr,
                                         kbuf, NN, Dd, Dd, 0);
    wmma_gemm<<<gD, blk256, 0, stream>>>(lnx, Wv + (size_t)l * Dd * Dd, nullptr,
                                         vbuf, NN, Dd, Dd, 0);
    for (int e = 0; e < Ee; e++) {
      wmma_gemm<<<gD, blk256, 0, stream>>>(lnx, Wq + ((size_t)l * Ee + e) * Dd * Dd,
                                           nullptr, qbuf, NN, Dd, Dd, 0);
      dim3 gA(Tt / 64, Bb * Hh);
      moa_attn_k<<<gA, blk128, 0, stream>>>(qbuf, kbuf, vbuf, obuf);
      wmma_gemm<<<gD, blk256, 0, stream>>>(obuf, Wo + ((size_t)l * Ee + e) * Dd * Dd,
                                           nullptr, tmp, NN, Dd, Dd, 0);
      scale_add_k<<<(NN * Dd + 255) / 256, blk256, 0, stream>>>(x, tmp, coef, e);
    }
    layernorm_k<<<NN, blk256, 0, stream>>>(x, ln2_g + l * Dd, ln2_b + l * Dd, lnx);
    dim3 gF((FF + BN - 1) / BN, (NN + BM - 1) / BM);
    wmma_gemm<<<gF, blk256, 0, stream>>>(lnx, fc_w + (size_t)l * Dd * FF,
                                         fc_b + (size_t)l * FF, tmp,
                                         NN, FF, Dd, FLAG_BIAS | FLAG_GELU);
    dim3 gP((Dd + BN - 1) / BN, (NN + BM - 1) / BM);
    wmma_gemm<<<gP, blk256, 0, stream>>>(tmp, proj_w + (size_t)l * FF * Dd,
                                         proj_b + (size_t)l * Dd, x,
                                         NN, Dd, FF, FLAG_BIAS | FLAG_ACC);
  }
  layernorm_k<<<NN, blk256, 0, stream>>>(x, lnf_g, lnf_b, lnx);
  dim3 gL((Vv + BN - 1) / BN, (NN + BM - 1) / BM);
  wmma_gemm<<<gL, blk256, 0, stream>>>(lnx, wte, nullptr, out, NN, Vv, Dd, FLAG_TB);
}
